// TemporalBand_56762287784389
// MI455X (gfx1250) — matrix operand order repaired and checked by reading, not compiled, not run on hardware
//
#include <hip/hip_runtime.h>
#include <hip/hip_bf16.h>

// ---------------------------------------------------------------------------
// Types / WMMA helpers (CDNA5 gfx1250, wave32)
// ---------------------------------------------------------------------------
typedef __bf16 bf16_t;
typedef __attribute__((ext_vector_type(16))) __bf16 v16bf;
typedef __attribute__((ext_vector_type(8)))  float  v8f;

__device__ __forceinline__ v8f wmma_bf16(v16bf a, v16bf b, v8f c) {
  // D = A(16x32 bf16) x B(32x16 bf16) + C(16x16 f32)
  return __builtin_amdgcn_wmma_f32_16x16x32_bf16(
      /*neg_a=*/false, a, /*neg_b=*/false, b,
      /*c_mod=*/(short)0, c, /*reuse_a=*/false, /*reuse_b=*/false);
}

// float -> bf16 round-to-nearest-even via integer ops
__device__ __forceinline__ bf16_t f2bf(float f) {
  unsigned u = __builtin_bit_cast(unsigned, f);
  unsigned r = (u + 0x7FFFu + ((u >> 16) & 1u)) >> 16;
  unsigned short s = (unsigned short)r;
  return __builtin_bit_cast(bf16_t, s);
}

// build a 16-element bf16 fragment from two 16B-aligned 8-element chunks
union BF16x16 { uint4 u[2]; v16bf v; };
__device__ __forceinline__ v16bf ld_frag16(const bf16_t* p0, const bf16_t* p1) {
  BF16x16 c;
  c.u[0] = *(const uint4*)p0;
  c.u[1] = *(const uint4*)p1;
  return c.v;
}

// ---------------------------------------------------------------------------
// Problem constants
// ---------------------------------------------------------------------------
#define BATCH   2
#define T_SEQ   2048
#define D_MODEL 1024
#define N_HEADS 16
#define HD      64

// ---------------------------------------------------------------------------
// Utility kernels
// ---------------------------------------------------------------------------
__global__ void zero_f32(float* __restrict__ p, size_t n) {
  size_t i = (size_t)blockIdx.x * blockDim.x + threadIdx.x;
  if (i < n) p[i] = 0.0f;
}

__global__ void cvt_f32_bf16(const float* __restrict__ src,
                             bf16_t* __restrict__ dst, size_t n) {
  size_t i = ((size_t)blockIdx.x * blockDim.x + threadIdx.x) * 4;
  if (i + 3 < n) {
    float4 v = *(const float4*)(src + i);
    dst[i + 0] = f2bf(v.x);
    dst[i + 1] = f2bf(v.y);
    dst[i + 2] = f2bf(v.z);
    dst[i + 3] = f2bf(v.w);
  } else {
    for (size_t j = i; j < n; ++j) dst[j] = f2bf(src[j]);
  }
}

// ---------------------------------------------------------------------------
// GEMM: C(MxN) = A(MxK bf16, row-major) * W(NxK bf16, row-major)^T + bias
// Block = 128 threads (4 waves); tile 64(M) x 64(N); wave w -> 16x64 strip.
// W tile is staged TRANSPOSED (wT[k][n], padded rows); B-fragments are
// gathered as 16 scalar LDS reads (stride 72). A fragments are ds_load_b128 pairs.
// ---------------------------------------------------------------------------
template <int OUT_BF16>
__global__ __launch_bounds__(128)
void gemm_awT(const bf16_t* __restrict__ A, const bf16_t* __restrict__ W,
              const float* __restrict__ bias,
              bf16_t* __restrict__ Cb, float* __restrict__ Cf,
              int M, int N, int K) {
  __shared__ __attribute__((aligned(16))) bf16_t aTile[64 * 32];
  __shared__ __attribute__((aligned(16))) bf16_t wT[32 * 72]; // [k][n], 144B rows

  const int tid  = threadIdx.x;
  const int lane = tid & 31;
  const int wave = tid >> 5;
  const int g    = lane >> 4;
  const int hl   = lane & 15;

  const int mbase = blockIdx.x * 64;
  const int nbase = blockIdx.y * 64;

  v8f acc[4] = {};

  for (int k0 = 0; k0 < K; k0 += 32) {
    {
      int row = tid >> 1;           // 0..63 : m-row for A, n-row for W
      int col = (tid & 1) * 16;     // k-chunk
      // A tile: row-major copy (b128 in, b128 out)
      const uint4* sa = (const uint4*)(A + (size_t)(mbase + row) * K + k0 + col);
      uint4* da = (uint4*)(aTile + row * 32 + col);
      da[0] = sa[0]; da[1] = sa[1];
      // W tile: transpose on store -> wT[k][n] = W[nbase+n][k0+k]
      union { uint4 u[2]; bf16_t e[16]; } wld;
      const uint4* sw = (const uint4*)(W + (size_t)(nbase + row) * K + k0 + col);
      wld.u[0] = sw[0]; wld.u[1] = sw[1];
#pragma unroll
      for (int e = 0; e < 16; ++e)
        wT[(col + e) * 72 + row] = wld.e[e];
    }
    __syncthreads();

    const bf16_t* ar = aTile + (wave * 16 + hl) * 32;
    v16bf af = ld_frag16(ar + g * 8, ar + 16 + g * 8);

#pragma unroll
    for (int nb = 0; nb < 4; ++nb) {
      v16bf bfr;  // B[k(e,g)][n=hl]
#pragma unroll
      for (int e = 0; e < 16; ++e)
        bfr[e] = wT[((e < 8) ? g * 8 + e : 8 + g * 8 + e) * 72 + nb * 16 + hl];
      acc[nb] = wmma_bf16(af, bfr, acc[nb]);
    }
    __syncthreads();
  }

#pragma unroll
  for (int nb = 0; nb < 4; ++nb) {
#pragma unroll
    for (int vr = 0; vr < 8; ++vr) {
      int row = mbase + wave * 16 + vr + 8 * g;
      int col = nbase + nb * 16 + hl;
      float v = acc[nb][vr] + bias[col];
      if (OUT_BF16) Cb[(size_t)row * N + col] = f2bf(v);
      else          Cf[(size_t)row * N + col] = v;
    }
  }
}

// ---------------------------------------------------------------------------
// Fused causal flash attention, S^T formulation: S^T = K * Q^T.
//  - K operand is the A-matrix -> fragments load DIRECTLY from global (rows
//    contiguous), no K staging, no LDS gather.
//  - Q^T B-fragments built once per wave through LDS.
//  - lane owns query q = lane&15 -> softmax stats are per-lane scalars;
//    2 shuffles per key-step (max & sum across the two half-waves).
// Pass 2 recomputes scores and atomically accumulates normalized probs / H
// into attn_avg[B,T,T] for the entropy reduction.
// ---------------------------------------------------------------------------
__global__ __launch_bounds__(128)
void attn_fused(const bf16_t* __restrict__ Q, const bf16_t* __restrict__ Km,
                const bf16_t* __restrict__ V, bf16_t* __restrict__ ctx,
                float* __restrict__ attn_avg) {
  __shared__ __attribute__((aligned(16))) bf16_t pls[4][16 * 72]; // 144B rows

  const int tid  = threadIdx.x;
  const int lane = tid & 31;
  const int wave = tid >> 5;
  const int g    = lane >> 4;
  const int hl   = lane & 15;

  const int QB  = T_SEQ / 16;
  const int wid = blockIdx.x * 4 + wave;
  const int qb  = wid % QB;
  const int h   = (wid / QB) % N_HEADS;
  const int b   = wid / (QB * N_HEADS);
  const int qbase = qb * 16;

  bf16_t* pbuf = pls[wave];

  // ---- one-time: B-fragments of Q^T  (B[d][q] = Q[qbase+q][d]) ----
  {
    const uint4* src = (const uint4*)(Q + (size_t)(b * T_SEQ + qbase + hl) * D_MODEL +
                                      h * HD + g * 32);
    uint4* dst = (uint4*)(pbuf + hl * 72 + g * 32);
#pragma unroll
    for (int i = 0; i < 4; ++i) dst[i] = src[i];
  }
  v16bf bq[2];
#pragma unroll
  for (int d0 = 0; d0 < 2; ++d0)
#pragma unroll
    for (int e = 0; e < 16; ++e)
      bq[d0][e] = pbuf[hl * 72 + d0 * 32 + ((e < 8) ? g * 8 + e : 8 + g * 8 + e)];

  const float scale = 0.125f;       // 1/sqrt(64)
  const int   q     = qbase + hl;   // this lane's query
  float mq = -1e30f, lq = 0.0f;
  v8f o[4] = {};

  const int kend = qbase + 16;      // exclusive causal key bound

  // ---------------- pass 1: flash attention ----------------
  for (int kb = 0; kb < kend; kb += 32) {
    // K A-fragments, direct from global (2 x b128 per fragment)
    v16bf ak[2][2];
#pragma unroll
    for (int t = 0; t < 2; ++t) {
      const bf16_t* kr = Km + (size_t)(b * T_SEQ + kb + 16 * t + hl) * D_MODEL + h * HD;
#pragma unroll
      for (int d0 = 0; d0 < 2; ++d0)
        ak[t][d0] = ld_frag16(kr + d0 * 32 + g * 8, kr + d0 * 32 + 16 + g * 8);
    }
    if (kb + 32 < kend)
      __builtin_prefetch(Km + (size_t)(b * T_SEQ + kb + 32 + lane) * D_MODEL + h * HD, 0, 1);

    // S^T tiles: rows = keys, cols = queries
    v8f st0 = {}, st1 = {};
    st0 = wmma_bf16(ak[0][0], bq[0], st0);
    st0 = wmma_bf16(ak[0][1], bq[1], st0);
    st1 = wmma_bf16(ak[1][0], bq[0], st1);
    st1 = wmma_bf16(ak[1][1], bq[1], st1);

    // masked, scaled scores; per-lane online softmax for query q
    float v0[8], v1[8];
    float mx = -1e30f;
#pragma unroll
    for (int vr = 0; vr < 8; ++vr) {
      int k0 = kb + vr + 8 * g;
      v0[vr] = (k0      <= q) ? st0[vr] * scale : -1e30f;
      v1[vr] = (k0 + 16 <= q) ? st1[vr] * scale : -1e30f;
      mx = fmaxf(mx, fmaxf(v0[vr], v1[vr]));
    }
    mx = fmaxf(mx, __shfl_xor(mx, 16, 32));   // combine the two half-waves
    float mn   = fmaxf(mq, mx);
    float corr = __expf(mq - mn);
    float rs = 0.0f;
#pragma unroll
    for (int vr = 0; vr < 8; ++vr) {
      v0[vr] = __expf(v0[vr] - mn);
      v1[vr] = __expf(v1[vr] - mn);
      rs += v0[vr] + v1[vr];
    }
    rs += __shfl_xor(rs, 16, 32);
    lq = lq * corr + rs;
    mq = mn;

    // rescale O: row vr of the O fragment is query qbase+vr+8*g
    float corrRow[8];
#pragma unroll
    for (int vr = 0; vr < 8; ++vr) corrRow[vr] = __shfl(corr, vr + 8 * g, 32);
#pragma unroll
    for (int nb = 0; nb < 4; ++nb)
#pragma unroll
      for (int vr = 0; vr < 8; ++vr) o[nb][vr] *= corrRow[vr];

    // P rows (q-major) to LDS: lane owns query hl, keys 16t+vr+8g
#pragma unroll
    for (int vr = 0; vr < 8; ++vr) {
      pbuf[hl * 72 + vr + 8 * g]      = f2bf(v0[vr]);
      pbuf[hl * 72 + 16 + vr + 8 * g] = f2bf(v1[vr]);
    }
    // A-fragment of P (16q x 32key): contiguous b128 pair
    const bf16_t* pr = pbuf + hl * 72;
    v16bf pf = ld_frag16(pr + g * 8, pr + 16 + g * 8);

    // O += P * V  (V B-fragment gathered: column hl, keys in table order)
#pragma unroll
    for (int nb = 0; nb < 4; ++nb) {
      v16bf bv;
#pragma unroll
      for (int e = 0; e < 16; ++e)
        bv[e] = V[(size_t)(b * T_SEQ + kb + ((e < 8) ? g * 8 + e : 8 + g * 8 + e)) * D_MODEL + h * HD + nb * 16 + hl];
      o[nb] = wmma_bf16(pf, bv, o[nb]);
    }
  }

  float invlq = 1.0f / lq;
  float invRow[8];
#pragma unroll
  for (int vr = 0; vr < 8; ++vr) invRow[vr] = __shfl(invlq, vr + 8 * g, 32);

  // ---------------- pass 2: head-averaged probability map ----------------
  const float hinv = 1.0f / (float)N_HEADS;
  for (int kb = 0; kb < kend; kb += 32) {
    v16bf ak[2][2];
#pragma unroll
    for (int t = 0; t < 2; ++t) {
      const bf16_t* kr = Km + (size_t)(b * T_SEQ + kb + 16 * t + hl) * D_MODEL + h * HD;
#pragma unroll
      for (int d0 = 0; d0 < 2; ++d0)
        ak[t][d0] = ld_frag16(kr + d0 * 32 + g * 8, kr + d0 * 32 + 16 + g * 8);
    }
    v8f st0 = {}, st1 = {};
    st0 = wmma_bf16(ak[0][0], bq[0], st0);
    st0 = wmma_bf16(ak[0][1], bq[1], st0);
    st1 = wmma_bf16(ak[1][0], bq[0], st1);
    st1 = wmma_bf16(ak[1][1], bq[1], st1);

    float* dst = attn_avg + ((size_t)b * T_SEQ + q) * T_SEQ;
#pragma unroll
    for (int vr = 0; vr < 8; ++vr) {
      int k0 = kb + vr + 8 * g;
      if (k0      <= q) atomicAdd(dst + k0,      __expf(st0[vr] * scale - mq) * invlq * hinv);
      if (k0 + 16 <= q) atomicAdd(dst + k0 + 16, __expf(st1[vr] * scale - mq) * invlq * hinv);
    }
  }

  // ---------------- epilogue: normalized context (bf16) ----------------
#pragma unroll
  for (int nb = 0; nb < 4; ++nb)
#pragma unroll
    for (int vr = 0; vr < 8; ++vr) {
      int row = qbase + vr + 8 * g;
      ctx[(size_t)(b * T_SEQ + row) * D_MODEL + h * HD + nb * 16 + hl] =
          f2bf(o[nb][vr] * invRow[vr]);
    }
}

// ---------------------------------------------------------------------------
// Entropy of head-averaged attention: one block per (b, q) row.
// ---------------------------------------------------------------------------
__global__ __launch_bounds__(256)
void entropy_kernel(const float* __restrict__ attn_avg, float* __restrict__ ent) {
  const int b = blockIdx.x / T_SEQ;
  const int q = blockIdx.x % T_SEQ;
  const float* row = attn_avg + ((size_t)b * T_SEQ + q) * T_SEQ;
  float s = 0.0f;
  for (int k = threadIdx.x; k <= q; k += 256) {
    float p = row[k];
    s -= p * __logf(p + 1e-9f);
  }
  __shared__ float red[256];
  red[threadIdx.x] = s;
  __syncthreads();
  for (int st = 128; st > 0; st >>= 1) {
    if (threadIdx.x < st) red[threadIdx.x] += red[threadIdx.x + st];
    __syncthreads();
  }
  if (threadIdx.x == 0) atomicAdd(&ent[b], red[0] * (1.0f / (float)T_SEQ));
}

// ---------------------------------------------------------------------------
// Host-side launch
// ---------------------------------------------------------------------------
extern "C" void kernel_launch(void* const* d_in, const int* in_sizes, int n_in,
                              void* d_out, int out_size, void* d_ws, size_t ws_size,
                              hipStream_t stream) {
  (void)in_sizes; (void)n_in; (void)out_size; (void)ws_size;

  const float* x  = (const float*)d_in[0];
  const float* Wq = (const float*)d_in[1];
  const float* bq = (const float*)d_in[2];
  const float* Wk = (const float*)d_in[3];
  const float* bk = (const float*)d_in[4];
  const float* Wv = (const float*)d_in[5];
  const float* bv = (const float*)d_in[6];
  const float* Wo = (const float*)d_in[7];
  const float* bo = (const float*)d_in[8];

  const size_t M   = (size_t)BATCH * T_SEQ;          // 4096
  const size_t BTD = M * D_MODEL;                    // 4,194,304
  const size_t WSZ = (size_t)D_MODEL * D_MODEL;      // 1,048,576
  const size_t BTT = (size_t)BATCH * T_SEQ * T_SEQ;  // 8,388,608

  float* out = (float*)d_out;
  float* ent = out + BTD;

  char* ws = (char*)d_ws;
  bf16_t* xb  = (bf16_t*)ws;              ws += BTD * sizeof(bf16_t);
  bf16_t* Wqb = (bf16_t*)ws;              ws += WSZ * sizeof(bf16_t);
  bf16_t* Wkb = (bf16_t*)ws;              ws += WSZ * sizeof(bf16_t);
  bf16_t* Wvb = (bf16_t*)ws;              ws += WSZ * sizeof(bf16_t);
  bf16_t* Wob = (bf16_t*)ws;              ws += WSZ * sizeof(bf16_t);
  bf16_t* Qb  = (bf16_t*)ws;              ws += BTD * sizeof(bf16_t);
  bf16_t* Kb  = (bf16_t*)ws;              ws += BTD * sizeof(bf16_t);
  bf16_t* Vb  = (bf16_t*)ws;              ws += BTD * sizeof(bf16_t);
  bf16_t* ctx = (bf16_t*)ws;              ws += BTD * sizeof(bf16_t);
  float*  attn_avg = (float*)ws;          ws += BTT * sizeof(float);

  zero_f32<<<dim3((unsigned)((BTT + 255) / 256)), dim3(256), 0, stream>>>(attn_avg, BTT);
  zero_f32<<<dim3(1), dim3(32), 0, stream>>>(ent, (size_t)BATCH);

  cvt_f32_bf16<<<dim3((unsigned)((BTD / 4 + 255) / 256)), dim3(256), 0, stream>>>(x,  xb,  BTD);
  cvt_f32_bf16<<<dim3((unsigned)((WSZ / 4 + 255) / 256)), dim3(256), 0, stream>>>(Wq, Wqb, WSZ);
  cvt_f32_bf16<<<dim3((unsigned)((WSZ / 4 + 255) / 256)), dim3(256), 0, stream>>>(Wk, Wkb, WSZ);
  cvt_f32_bf16<<<dim3((unsigned)((WSZ / 4 + 255) / 256)), dim3(256), 0, stream>>>(Wv, Wvb, WSZ);
  cvt_f32_bf16<<<dim3((unsigned)((WSZ / 4 + 255) / 256)), dim3(256), 0, stream>>>(Wo, Wob, WSZ);

  dim3 ggrid((unsigned)(M / 64), D_MODEL / 64);
  gemm_awT<1><<<ggrid, dim3(128), 0, stream>>>(xb, Wqb, bq, Qb, nullptr,
                                               (int)M, D_MODEL, D_MODEL);
  gemm_awT<1><<<ggrid, dim3(128), 0, stream>>>(xb, Wkb, bk, Kb, nullptr,
                                               (int)M, D_MODEL, D_MODEL);
  gemm_awT<1><<<ggrid, dim3(128), 0, stream>>>(xb, Wvb, bv, Vb, nullptr,
                                               (int)M, D_MODEL, D_MODEL);

  const unsigned attn_waves = (unsigned)(BATCH * N_HEADS * (T_SEQ / 16)); // 4096
  attn_fused<<<dim3(attn_waves / 4), dim3(128), 0, stream>>>(Qb, Kb, Vb, ctx,
                                                             attn_avg);

  entropy_kernel<<<dim3((unsigned)(BATCH * T_SEQ)), dim3(256), 0, stream>>>(
      attn_avg, ent);

  gemm_awT<0><<<ggrid, dim3(128), 0, stream>>>(ctx, Wob, bo, nullptr, out,
                                               (int)M, D_MODEL, D_MODEL);
}
